// SA_18262200943196
// MI455X (gfx1250) — matrix operand order repaired and checked by reading, not compiled, not run on hardware
//
#include <hip/hip_runtime.h>

typedef unsigned short u16;
typedef unsigned int   u32;
typedef __attribute__((ext_vector_type(16))) __bf16 v16bf;
typedef __attribute__((ext_vector_type(8)))  float  v8f;

#define B_   8
#define C_   256
#define HW_  1024
#define NTOT (B_ * HW_)          // 8192
#define TM   128
#define TN   128
#define KC   64                  // K staged per LDS round (4 rounds of K=256)
#define LDS_STRIDE 72            // tile row stride (elements, padded, 16B aligned)

union FragAB { v16bf bf; uint4 q[2]; };

// ---- helpers -------------------------------------------------------------
__device__ __forceinline__ u16 f2bf(float f) {      // f32 -> bf16 RNE
  u32 u = __float_as_uint(f);
  u32 r = u + 0x7FFFu + ((u >> 16) & 1u);
  return (u16)(r >> 16);
}
__device__ __forceinline__ u32 fkey(float f) {      // monotonic uint key for float max
  u32 b = __float_as_uint(f);
  return (b & 0x80000000u) ? ~b : (b | 0x80000000u);
}
__device__ __forceinline__ float keyf(u32 k) {
  u32 b = (k & 0x80000000u) ? (k & 0x7FFFFFFFu) : ~k;
  return __uint_as_float(b);
}

// ---- CDNA5 async global->LDS copy (ASYNCcnt-tracked, no VGPR round-trip) --
// GVS form: lds byte addr in VGPR (vdst), 32-bit byte offset in VGPR (vaddr),
// 64-bit base in SGPR pair (saddr).  LDS address = low 32 bits of the generic
// shared-aperture pointer (ISA 10.2 aperture truncation).
__device__ __forceinline__ void async_copy_b128(u16* lds_dst,
                                                const u16* __restrict__ base,
                                                u32 elem_off) {
  u32 laddr = (u32)(size_t)lds_dst;
  u32 boff  = elem_off * 2u;
  asm volatile("global_load_async_to_lds_b128 %0, %1, %2"
               :: "v"(laddr), "v"(boff), "s"(base)
               : "memory");
}
__device__ __forceinline__ void wait_async() {
  asm volatile("s_wait_asynccnt 0x0" ::: "memory");
}

// ---- shared WMMA mainloop ------------------------------------------------
// C(128x128) += A(128x256, row-major-in-K) * B(128 cols x 256 K, ALSO stored
// row-major-in-K: Bt[n*256 + k]).  Both tiles stage identically via async
// global->LDS b128 copies; all fragment loads are aligned ds_load_b128.
__device__ __forceinline__ void gemm_mainloop(
    const u16* __restrict__ A,   // A  + gm0*256   : A [m*256 + k]
    const u16* __restrict__ Bt,  // Bt + n0*256    : Bt[n*256 + k]
    u16* As, u16* Bs, v8f acc[4][2])
{
  const int tid  = threadIdx.x;
  const int lane = tid & 31;
  const int w    = tid >> 5;
  const int wr   = w >> 2;       // 0..1  (64-row slab)
  const int wc   = w & 3;        // 0..3  (32-col slab)
  const int half = lane >> 4;
  const int l15  = lane & 15;

  for (int st = 0; st < 4; ++st) {
    const int kb = st * KC;
    // stage A and B tiles: each 128 rows x 64 K bf16 (1024 x 16B chunks),
    // DMA'd straight into LDS (async, ASYNCcnt)
#pragma unroll
    for (int j = 0; j < 4; ++j) {
      int i  = tid + 256 * j;
      int m  = i >> 3;
      int c8 = (i & 7) * 8;
      async_copy_b128(&As[m * LDS_STRIDE + c8], A,  (u32)(m * 256 + kb + c8));
      async_copy_b128(&Bs[m * LDS_STRIDE + c8], Bt, (u32)(m * 256 + kb + c8));
    }
    if (st + 1 < 4) {   // warm L2 for next stage -> global_prefetch_b8
      __builtin_prefetch(&A [(tid >> 3) * 256 + kb + KC + (tid & 7) * 8], 0, 1);
      __builtin_prefetch(&Bt[(tid >> 3) * 256 + kb + KC + (tid & 7) * 8], 0, 1);
    }
    wait_async();          // this wave's LDS writes landed
    __syncthreads();       // all waves' writes visible

#pragma unroll
    for (int kk = 0; kk < KC; kk += 32) {
      FragAB a[4];
#pragma unroll
      for (int mt = 0; mt < 4; ++mt) {
        int row = wr * 64 + mt * 16 + l15;
        // lanes<16: K 0..7 & 16..23 ; lanes>=16: K 8..15 & 24..31
        a[mt].q[0] = *(const uint4*)&As[row * LDS_STRIDE + kk + half * 8];
        a[mt].q[1] = *(const uint4*)&As[row * LDS_STRIDE + kk + 16 + half * 8];
      }
      FragAB b[2];
#pragma unroll
      for (int nt = 0; nt < 2; ++nt) {
        int n = wc * 32 + nt * 16 + l15;    // B: N=l15; lanes<16 K0..7 & 16..23, else K8..15 & 24..31
        b[nt].q[0] = *(const uint4*)&Bs[n * LDS_STRIDE + kk + half * 8];
        b[nt].q[1] = *(const uint4*)&Bs[n * LDS_STRIDE + kk + 16 + half * 8];
      }
#pragma unroll
      for (int mt = 0; mt < 4; ++mt)
#pragma unroll
        for (int nt = 0; nt < 2; ++nt)
          acc[mt][nt] = __builtin_amdgcn_wmma_f32_16x16x32_bf16(
              false, a[mt].bf, false, b[nt].bf, (short)0, acc[mt][nt],
              false, false);
    }
    __syncthreads();
  }
}

// ---- k0: convert x / weights to bf16, init reduction buffers -------------
__global__ __launch_bounds__(256) void init_kernel(
    const float* __restrict__ x, const float* __restrict__ Wq,
    const float* __restrict__ Wk, const float* __restrict__ Wo,
    u16* __restrict__ xbfT, u16* __restrict__ wqbf, u16* __restrict__ wkbf,
    u16* __restrict__ wobf, u32* __restrict__ maxKey, float* __restrict__ rowSum)
{
  int i = blockIdx.x * 256 + threadIdx.x;
  if (i < B_ * C_ * HW_) {
    int b = i >> 18;            // / (C_*HW_)
    int c = (i >> 10) & 255;    // channel
    int s = i & 1023;           // spatial
    xbfT[(size_t)b * (HW_ * C_) + s * C_ + c] = f2bf(x[i]);  // coalesced read
  }
  if (i < C_ * C_) {
    wqbf[i] = f2bf(Wq[i]);
    wkbf[i] = f2bf(Wk[i]);
    wobf[i] = f2bf(Wo[i]);
  }
  if (i < NTOT * B_) { maxKey[i] = 0u; rowSum[i] = 0.0f; }
}

// ---- k1: projection GEMM: O = W(256x256) * Xt^T + bias -------------------
// TRANSPOSE_OUT=false : Obf[b][o][s]  (Q -> A operand of attention GEMM)
// TRANSPOSE_OUT=true  : Obf[b][s][o]  (K -> K-major B operand)
template <bool TRANSPOSE_OUT>
__global__ __launch_bounds__(256) void proj_gemm_kernel(
    const u16* __restrict__ Wbf, const float* __restrict__ bias,
    const u16* __restrict__ XbfT, u16* __restrict__ Obf)
{
  __shared__ u16 As[TM * LDS_STRIDE];
  __shared__ u16 Bs[TN * LDS_STRIDE];
  const int n0  = blockIdx.x * TN;     // spatial tile
  const int gm0 = blockIdx.y * TM;     // out-channel tile
  const int bat = blockIdx.z;
  v8f acc[4][2];
#pragma unroll
  for (int mt = 0; mt < 4; ++mt)
#pragma unroll
    for (int nt = 0; nt < 2; ++nt)
#pragma unroll
      for (int v = 0; v < 8; ++v) acc[mt][nt][v] = 0.0f;

  gemm_mainloop(Wbf + gm0 * 256,
                XbfT + (size_t)bat * (HW_ * C_) + n0 * C_, As, Bs, acc);

  const int tid = threadIdx.x, lane = tid & 31, w = tid >> 5;
  const int wr = w >> 2, wc = w & 3, half = lane >> 4, l15 = lane & 15;
#pragma unroll
  for (int mt = 0; mt < 4; ++mt)
#pragma unroll
    for (int nt = 0; nt < 2; ++nt)
#pragma unroll
      for (int v = 0; v < 8; ++v) {
        int row = gm0 + wr * 64 + mt * 16 + half * 8 + v;   // out channel o
        int col = n0 + wc * 32 + nt * 16 + l15;             // spatial s
        u16 val = f2bf(acc[mt][nt][v] + bias[row]);
        if (TRANSPOSE_OUT)
          Obf[(size_t)bat * (C_ * HW_) + col * C_ + row] = val;
        else
          Obf[(size_t)bat * (C_ * HW_) + row * HW_ + col] = val;
      }
}

// ---- k2: S = q@k fused with per-(row, batch-block) max/sum reduction -----
__global__ __launch_bounds__(256) void attn_gemm_kernel(
    const u16* __restrict__ Qbf, const u16* __restrict__ KbfT,
    u32* __restrict__ maxKey, float* __restrict__ rowSum)
{
  __shared__ u16 As[TM * LDS_STRIDE];
  __shared__ u16 Bs[TN * LDS_STRIDE];
  __shared__ float redM[TM * 4];
  __shared__ float redS[TM * 4];
  const int tileN = blockIdx.x;            // 0..63 -> col block of 128
  const int tileM = blockIdx.y;            // 0..63 -> row block of 128
  const int bcol  = tileN >> 3;            // batch block b' (1024 cols each)
  const int s0    = (tileN & 7) * TN;
  const int gm0   = tileM * TM;

  v8f acc[4][2];
#pragma unroll
  for (int mt = 0; mt < 4; ++mt)
#pragma unroll
    for (int nt = 0; nt < 2; ++nt)
#pragma unroll
      for (int v = 0; v < 8; ++v) acc[mt][nt][v] = 0.0f;

  // A: flat-reshaped Q [8192 x 256]; B: KbfT[b'][s'][c] -> Bt[n*256 + k]
  gemm_mainloop(Qbf + (size_t)gm0 * 256,
                KbfT + (size_t)bcol * (C_ * HW_) + s0 * C_, As, Bs, acc);

  const int tid = threadIdx.x, lane = tid & 31, w = tid >> 5;
  const int wr = w >> 2, wc = w & 3, half = lane >> 4, l15 = lane & 15;
#pragma unroll
  for (int mt = 0; mt < 4; ++mt) {
    float mx[8], sm[8];
#pragma unroll
    for (int v = 0; v < 8; ++v) {
      mx[v] = fmaxf(acc[mt][0][v], acc[mt][1][v]);
      sm[v] = acc[mt][0][v] + acc[mt][1][v];
    }
#pragma unroll
    for (int off = 1; off < 16; off <<= 1) {   // reduce across 16 lanes of each half
#pragma unroll
      for (int v = 0; v < 8; ++v) {
        mx[v] = fmaxf(mx[v], __shfl_xor(mx[v], off, 32));
        sm[v] += __shfl_xor(sm[v], off, 32);
      }
    }
    if (l15 == 0) {
#pragma unroll
      for (int v = 0; v < 8; ++v) {
        int row = wr * 64 + mt * 16 + half * 8 + v;
        redM[row * 4 + wc] = mx[v];
        redS[row * 4 + wc] = sm[v];
      }
    }
  }
  __syncthreads();
  if (tid < TM) {
    float m = fmaxf(fmaxf(redM[tid * 4 + 0], redM[tid * 4 + 1]),
                    fmaxf(redM[tid * 4 + 2], redM[tid * 4 + 3]));
    float s = redS[tid * 4 + 0] + redS[tid * 4 + 1] +
              redS[tid * 4 + 2] + redS[tid * 4 + 3];
    int r = gm0 + tid;
    atomicMax(&maxKey[r * 8 + bcol], fkey(m));
    atomicAdd(&rowSum[r * 8 + bcol], s);
  }
}

// ---- k3: x_co = softmax over HW of scale * mean_b'(max + mean) -----------
__global__ __launch_bounds__(256) void softmax_kernel(
    const u32* __restrict__ maxKey, const float* __restrict__ rowSum,
    float* __restrict__ xco)
{
  __shared__ float sred[256];
  const int bb = blockIdx.x, tid = threadIdx.x;
  float v[4];
#pragma unroll
  for (int j = 0; j < 4; ++j) {
    int r = bb * 1024 + tid + 256 * j;
    float a = 0.0f;
#pragma unroll
    for (int b2 = 0; b2 < 8; ++b2)
      a += keyf(maxKey[r * 8 + b2]) + rowSum[r * 8 + b2] * (1.0f / 1024.0f);
    v[j] = a * (1.0f / 8.0f) * 0.0625f;          // mean over batch * 1/sqrt(C)
  }
  float lm = fmaxf(fmaxf(v[0], v[1]), fmaxf(v[2], v[3]));
  sred[tid] = lm; __syncthreads();
  for (int s = 128; s > 0; s >>= 1) {
    if (tid < s) sred[tid] = fmaxf(sred[tid], sred[tid + s]);
    __syncthreads();
  }
  float mx = sred[0]; __syncthreads();
  float e[4], ls = 0.0f;
#pragma unroll
  for (int j = 0; j < 4; ++j) { e[j] = __expf(v[j] - mx); ls += e[j]; }
  sred[tid] = ls; __syncthreads();
  for (int s = 128; s > 0; s >>= 1) {
    if (tid < s) sred[tid] += sred[tid + s];
    __syncthreads();
  }
  float inv = 1.0f / sred[0];
#pragma unroll
  for (int j = 0; j < 4; ++j) xco[bb * 1024 + tid + 256 * j] = e[j] * inv;
}

// ---- k3b: xsT[b][s][c] = bf16(x[b][c][s] * x_co[b][s]) -------------------
__global__ __launch_bounds__(256) void scale_kernel(
    const float* __restrict__ x, const float* __restrict__ xco,
    u16* __restrict__ xsbfT)
{
  int i = blockIdx.x * 256 + threadIdx.x;       // input-linear: coalesced read
  int b = i >> 18;
  int c = (i >> 10) & 255;
  int s = i & 1023;
  xsbfT[(size_t)b * (HW_ * C_) + s * C_ + c] = f2bf(x[i] * xco[(b << 10) | s]);
}

// ---- k4: output conv GEMM, f32 output + bias -----------------------------
__global__ __launch_bounds__(256) void out_gemm_kernel(
    const u16* __restrict__ Wbf, const float* __restrict__ bias,
    const u16* __restrict__ XSbfT, float* __restrict__ Out)
{
  __shared__ u16 As[TM * LDS_STRIDE];
  __shared__ u16 Bs[TN * LDS_STRIDE];
  const int n0  = blockIdx.x * TN;
  const int gm0 = blockIdx.y * TM;
  const int bat = blockIdx.z;
  v8f acc[4][2];
#pragma unroll
  for (int mt = 0; mt < 4; ++mt)
#pragma unroll
    for (int nt = 0; nt < 2; ++nt)
#pragma unroll
      for (int v = 0; v < 8; ++v) acc[mt][nt][v] = 0.0f;

  gemm_mainloop(Wbf + gm0 * 256,
                XSbfT + (size_t)bat * (HW_ * C_) + n0 * C_, As, Bs, acc);

  const int tid = threadIdx.x, lane = tid & 31, w = tid >> 5;
  const int wr = w >> 2, wc = w & 3, half = lane >> 4, l15 = lane & 15;
#pragma unroll
  for (int mt = 0; mt < 4; ++mt)
#pragma unroll
    for (int nt = 0; nt < 2; ++nt)
#pragma unroll
      for (int v = 0; v < 8; ++v) {
        int row = gm0 + wr * 64 + mt * 16 + half * 8 + v;
        int col = n0 + wc * 32 + nt * 16 + l15;
        Out[(size_t)bat * (C_ * HW_) + row * HW_ + col] =
            acc[mt][nt][v] + bias[row];
      }
}

// ---- host ----------------------------------------------------------------
extern "C" void kernel_launch(void* const* d_in, const int* in_sizes, int n_in,
                              void* d_out, int out_size, void* d_ws, size_t ws_size,
                              hipStream_t stream)
{
  const float* x  = (const float*)d_in[0];
  const float* Wq = (const float*)d_in[1];
  const float* bq = (const float*)d_in[2];
  const float* Wk = (const float*)d_in[3];
  const float* bk = (const float*)d_in[4];
  const float* Wo = (const float*)d_in[5];
  const float* bo = (const float*)d_in[6];
  float* out = (float*)d_out;

  char* ws = (char*)d_ws;
  size_t off = 0;
  u16* xbfT  = (u16*)(ws + off); off += (size_t)B_ * C_ * HW_ * 2;   // 4 MB
  u16* Qbf   = (u16*)(ws + off); off += (size_t)B_ * C_ * HW_ * 2;   // 4 MB
  u16* KbfT  = (u16*)(ws + off); off += (size_t)B_ * C_ * HW_ * 2;   // 4 MB
  u16* XSbfT = (u16*)(ws + off); off += (size_t)B_ * C_ * HW_ * 2;   // 4 MB
  u16* wqbf  = (u16*)(ws + off); off += (size_t)C_ * C_ * 2;
  u16* wkbf  = (u16*)(ws + off); off += (size_t)C_ * C_ * 2;
  u16* wobf  = (u16*)(ws + off); off += (size_t)C_ * C_ * 2;
  u32* maxKey = (u32*)(ws + off); off += (size_t)NTOT * B_ * 4;
  float* rowSum = (float*)(ws + off); off += (size_t)NTOT * B_ * 4;
  float* xco = (float*)(ws + off); off += (size_t)NTOT * 4;

  init_kernel<<<(B_ * C_ * HW_) / 256, 256, 0, stream>>>(
      x, Wq, Wk, Wo, xbfT, wqbf, wkbf, wobf, maxKey, rowSum);

  dim3 gproj(HW_ / TN, C_ / TM, B_);   // (8, 2, 8)
  proj_gemm_kernel<false><<<gproj, 256, 0, stream>>>(wqbf, bq, xbfT, Qbf);   // Q
  proj_gemm_kernel<true ><<<gproj, 256, 0, stream>>>(wkbf, bk, xbfT, KbfT);  // K^T

  attn_gemm_kernel<<<dim3(NTOT / TN, NTOT / TM), 256, 0, stream>>>(
      Qbf, KbfT, maxKey, rowSum);

  softmax_kernel<<<B_, 256, 0, stream>>>(maxKey, rowSum, xco);
  scale_kernel<<<(B_ * C_ * HW_) / 256, 256, 0, stream>>>(x, xco, XSbfT);
  out_gemm_kernel<<<gproj, 256, 0, stream>>>(wobf, bo, XSbfT, out);
}
